// RelSO3OutBlockR_62337155334544
// MI455X (gfx1250) — matrix operand order repaired and checked by reading, not compiled, not run on hardware
//
#include <hip/hip_runtime.h>
#include <hip/hip_bf16.h>

// ---------------- problem constants ----------------
#define NBATCH   16
#define CCH      256        // channels
#define NPTS     256        // points (max-pooled)
#define NANCH    60         // anchors
#define NA_PAD   64         // padded anchors (4 tiles of 16)
#define KPAD     288        // 256 ch + 3 xyz, padded to 9*32
#define K1       512        // 2C input of first MLP layer
#define MLPD     256        // mlp width
#define OUTD     6

typedef __attribute__((ext_vector_type(16))) __bf16 v16bf;
typedef __attribute__((ext_vector_type(8)))  __bf16 v8bf;
typedef __attribute__((ext_vector_type(8)))  float  v8f;

// native conversions -> v_cvt_*bf16* instructions instead of integer emulation
static __device__ __forceinline__ __bf16 f2bf(float f) { return (__bf16)f; }
static __device__ __forceinline__ float  bf2f(__bf16 h) { return (float)h; }

// =====================================================================
// Kernel A: pointnet embed  g[cloud][b][o][a] = relu(max_p(Wp*feats)+bp)
// grid (32, 4): x = cloud*16+b, y = a-tile.  512 threads = 16 waves,
// wave w owns output-channel tile [16w, 16w+16).
// =====================================================================
__global__ __launch_bounds__(512)
void pointnet_wmma_kernel(const float* __restrict__ f1, const float* __restrict__ f2,
                          const float* __restrict__ x1, const float* __restrict__ x2,
                          const float* __restrict__ Wp, const float* __restrict__ bp,
                          float* __restrict__ g /* [2][NBATCH][CCH][NA_PAD] */) {
  const int bx    = blockIdx.x;
  const int cloud = bx >> 4;
  const int b     = bx & 15;
  const int a0    = blockIdx.y * 16;
  const bool full_tile = (a0 + 16 <= NANCH);          // tiles 0..2: no guards
  const float* __restrict__ f = cloud ? f2 : f1;
  const float* __restrict__ x = cloud ? x2 : x1;

  // transposed feature slice for one p: ft[a_local][k]  (B-operand layout)
  __shared__ __align__(64) __bf16 ft[16][KPAD];       // 9 KB

  const int t    = threadIdx.x;
  const int wave = t >> 5;
  const int lane = t & 31;
  const int ln   = lane & 15;
  const int hi   = lane >> 4;                         // 0 / 1
  const int m_base = wave * 16;

  // ---- cache all 9 A-tiles (bf16 Wp rows) in registers ----
  // 16-bit A layout: lane m = ln, k elements {kb..kb+7, kb+16..kb+23}, kb = 32*ch + 8*hi
  v16bf A[9];
  {
    const float* __restrict__ wrow = Wp + (size_t)(m_base + ln) * (CCH + 3);
    #pragma unroll
    for (int ch = 0; ch < 9; ++ch) {
      const int kb = ch * 32 + hi * 8;
      v16bf a;
      #pragma unroll
      for (int i = 0; i < 8; ++i) {
        int k0 = kb + i, k1 = kb + 16 + i;
        a[i]     = f2bf(k0 < CCH + 3 ? wrow[k0] : 0.0f);
        a[8 + i] = f2bf(k1 < CCH + 3 ? wrow[k1] : 0.0f);
      }
      A[ch] = a;
    }
  }

  v8f gmax;
  #pragma unroll
  for (int r = 0; r < 8; ++r) gmax[r] = -3.4e38f;

  for (int p = 0; p < NPTS; ++p) {
    __syncthreads();
    // ---- stage ft[j][k] = feats[k, p, a0+j] as bf16 (transposed) ----
    if (t < KPAD) {
      const int c = t;
      if (c < CCH) {
        const float* __restrict__ src =
            f + ((size_t)b * CCH + c) * ((size_t)NPTS * NANCH) + (size_t)p * NANCH + a0;
        if (p + 1 < NPTS) __builtin_prefetch(src + NANCH, 0, 1);  // global_prefetch
        if (full_tile) {
          // 16 contiguous, 16B-aligned floats -> 4 x global_load_b128
          const float4 v0 = ((const float4*)src)[0];
          const float4 v1 = ((const float4*)src)[1];
          const float4 v2 = ((const float4*)src)[2];
          const float4 v3 = ((const float4*)src)[3];
          ft[ 0][c] = f2bf(v0.x); ft[ 1][c] = f2bf(v0.y);
          ft[ 2][c] = f2bf(v0.z); ft[ 3][c] = f2bf(v0.w);
          ft[ 4][c] = f2bf(v1.x); ft[ 5][c] = f2bf(v1.y);
          ft[ 6][c] = f2bf(v1.z); ft[ 7][c] = f2bf(v1.w);
          ft[ 8][c] = f2bf(v2.x); ft[ 9][c] = f2bf(v2.y);
          ft[10][c] = f2bf(v2.z); ft[11][c] = f2bf(v2.w);
          ft[12][c] = f2bf(v3.x); ft[13][c] = f2bf(v3.y);
          ft[14][c] = f2bf(v3.z); ft[15][c] = f2bf(v3.w);
        } else {
          #pragma unroll
          for (int j = 0; j < 16; ++j)
            ft[j][c] = f2bf((a0 + j < NANCH) ? src[j] : 0.0f);
        }
      } else if (c < CCH + 3) {
        const float v = x[((size_t)b * NPTS + p) * 3 + (c - CCH)];
        #pragma unroll
        for (int j = 0; j < 16; ++j)
          ft[j][c] = f2bf((a0 + j < NANCH) ? v : 0.0f);
      } else {
        #pragma unroll
        for (int j = 0; j < 16; ++j) ft[j][c] = f2bf(0.0f);
      }
    }
    __syncthreads();

    // ---- acc = Wp_tile * feats_slice over K=288 ----
    v8f acc;
    #pragma unroll
    for (int r = 0; r < 8; ++r) acc[r] = 0.0f;
    #pragma unroll
    for (int ch = 0; ch < 9; ++ch) {
      const int kb = ch * 32 + hi * 8;                // B layout: k {kb..kb+7, kb+16..kb+23}
      v8bf lo = *(const v8bf*)&ft[ln][kb];
      v8bf hh = *(const v8bf*)&ft[ln][kb + 16];
      v16bf bm;
      #pragma unroll
      for (int i = 0; i < 8; ++i) { bm[i] = lo[i]; bm[8 + i] = hh[i]; }
      acc = __builtin_amdgcn_wmma_f32_16x16x32_bf16(false, A[ch], false, bm,
                                                    (short)0, acc, false, false);
    }
    #pragma unroll
    for (int r = 0; r < 8; ++r) gmax[r] = fmaxf(gmax[r], acc[r]);
  }

  // ---- bias + relu + store pooled features ----
  {
    const int a = a0 + ln;
    float* __restrict__ gout = g + ((size_t)cloud * NBATCH + b) * CCH * (size_t)NA_PAD;
    #pragma unroll
    for (int r = 0; r < 8; ++r) {
      const int m = m_base + hi * 8 + r;
      gout[(size_t)m * NA_PAD + a] = fmaxf(gmax[r] + bp[m], 0.0f);
    }
  }
}

// =====================================================================
// Kernel B: fused pairwise MLP + heads + softmax.
// grid = NBATCH*NANCH (one block per (b, a2)); 512 threads = 16 waves.
// Columns = a1 (padded to 64, 4 N-tiles). h0/h1 staged transposed in LDS.
// =====================================================================
__global__ __launch_bounds__(512)
void pairmlp_wmma_kernel(const float* __restrict__ g /* [2][NB][C][NA_PAD] */,
                         const float* __restrict__ W0, const float* __restrict__ b0,
                         const float* __restrict__ W1, const float* __restrict__ b1,
                         const float* __restrict__ Wa, const float* __restrict__ ba,
                         const float* __restrict__ Wr, const float* __restrict__ br,
                         float* __restrict__ out) {
  const int blk = blockIdx.x;
  const int b   = blk / NANCH;
  const int a2  = blk % NANCH;

  __shared__ __align__(64) __bf16 h0t[NA_PAD][MLPD];  // 32 KB, [a1][k]
  __shared__ __align__(64) __bf16 h1t[NA_PAD][MLPD];  // 32 KB
  __shared__ float attn_s[NA_PAD];

  const int t    = threadIdx.x;
  const int wave = t >> 5;
  const int lane = t & 31;
  const int ln   = lane & 15;
  const int hi   = lane >> 4;
  const int m_base = wave * 16;

  const float* __restrict__ g1 = g + ((size_t)0 * NBATCH + b) * CCH * (size_t)NA_PAD;
  const float* __restrict__ g2 = g + ((size_t)1 * NBATCH + b) * CCH * (size_t)NA_PAD;

  float bias0[8], bias1[8];
  #pragma unroll
  for (int r = 0; r < 8; ++r) {
    bias0[r] = b0[m_base + hi * 8 + r];
    bias1[r] = b1[m_base + hi * 8 + r];
  }

  // ---------------- phase 1: h0 = relu(W0 [g1(a1); g2(a2)] + b0) ----------------
  v8f acc[4];
  #pragma unroll
  for (int nt = 0; nt < 4; ++nt)
    #pragma unroll
    for (int r = 0; r < 8; ++r) acc[nt][r] = 0.0f;

  const float* __restrict__ w0row = W0 + (size_t)(m_base + ln) * K1;
  for (int ch = 0; ch < 16; ++ch) {
    const int kb = ch * 32 + hi * 8;                  // multiple of 8 -> 16B aligned
    if (ch + 1 < 16) __builtin_prefetch(w0row + kb + 32, 0, 1);
    const float4 wv0 = *(const float4*)(w0row + kb);
    const float4 wv1 = *(const float4*)(w0row + kb + 4);
    const float4 wv2 = *(const float4*)(w0row + kb + 16);
    const float4 wv3 = *(const float4*)(w0row + kb + 20);
    v16bf a;
    a[0]  = f2bf(wv0.x); a[1]  = f2bf(wv0.y); a[2]  = f2bf(wv0.z); a[3]  = f2bf(wv0.w);
    a[4]  = f2bf(wv1.x); a[5]  = f2bf(wv1.y); a[6]  = f2bf(wv1.z); a[7]  = f2bf(wv1.w);
    a[8]  = f2bf(wv2.x); a[9]  = f2bf(wv2.y); a[10] = f2bf(wv2.z); a[11] = f2bf(wv2.w);
    a[12] = f2bf(wv3.x); a[13] = f2bf(wv3.y); a[14] = f2bf(wv3.z); a[15] = f2bf(wv3.w);
    const int kb2 = ch * 32 + hi * 8;
    #pragma unroll
    for (int nt = 0; nt < 4; ++nt) {
      const int a1 = nt * 16 + ln;
      v16bf bm;
      if (ch < 8) {                                   // z[0:256) = g1[:, a1]
        #pragma unroll
        for (int i = 0; i < 16; ++i)
          bm[i] = f2bf(g1[(size_t)(kb2 + i + (i & 8)) * NA_PAD + a1]);
      } else {                                        // z[256:512) = g2[:, a2]
        #pragma unroll
        for (int i = 0; i < 16; ++i)
          bm[i] = f2bf(g2[(size_t)(kb2 - 256 + i + (i & 8)) * NA_PAD + a2]);
      }
      acc[nt] = __builtin_amdgcn_wmma_f32_16x16x32_bf16(false, a, false, bm,
                                                        (short)0, acc[nt], false, false);
    }
  }
  #pragma unroll
  for (int nt = 0; nt < 4; ++nt) {
    const int a1 = nt * 16 + ln;
    const int m0 = m_base + hi * 8;
    v8bf pk;
    #pragma unroll
    for (int r = 0; r < 8; ++r)
      pk[r] = f2bf(fmaxf(acc[nt][r] + bias0[r], 0.0f));
    *(v8bf*)&h0t[a1][m0] = pk;                        // ds_store_b128, transposed
  }
  __syncthreads();

  // ---------------- phase 2: h1 = relu(W1 h0 + b1) ----------------
  v8f acc2[4];
  #pragma unroll
  for (int nt = 0; nt < 4; ++nt)
    #pragma unroll
    for (int r = 0; r < 8; ++r) acc2[nt][r] = 0.0f;

  const float* __restrict__ w1row = W1 + (size_t)(m_base + ln) * MLPD;
  for (int ch = 0; ch < 8; ++ch) {
    const int kb = ch * 32 + hi * 8;
    const float4 wv0 = *(const float4*)(w1row + kb);
    const float4 wv1 = *(const float4*)(w1row + kb + 4);
    const float4 wv2 = *(const float4*)(w1row + kb + 16);
    const float4 wv3 = *(const float4*)(w1row + kb + 20);
    v16bf a;
    a[0]  = f2bf(wv0.x); a[1]  = f2bf(wv0.y); a[2]  = f2bf(wv0.z); a[3]  = f2bf(wv0.w);
    a[4]  = f2bf(wv1.x); a[5]  = f2bf(wv1.y); a[6]  = f2bf(wv1.z); a[7]  = f2bf(wv1.w);
    a[8]  = f2bf(wv2.x); a[9]  = f2bf(wv2.y); a[10] = f2bf(wv2.z); a[11] = f2bf(wv2.w);
    a[12] = f2bf(wv3.x); a[13] = f2bf(wv3.y); a[14] = f2bf(wv3.z); a[15] = f2bf(wv3.w);
    const int kb2 = ch * 32 + hi * 8;
    #pragma unroll
    for (int nt = 0; nt < 4; ++nt) {
      const int a1 = nt * 16 + ln;
      v8bf lo = *(const v8bf*)&h0t[a1][kb2];
      v8bf hh = *(const v8bf*)&h0t[a1][kb2 + 16];
      v16bf bm;
      #pragma unroll
      for (int i = 0; i < 8; ++i) { bm[i] = lo[i]; bm[8 + i] = hh[i]; }
      acc2[nt] = __builtin_amdgcn_wmma_f32_16x16x32_bf16(false, a, false, bm,
                                                         (short)0, acc2[nt], false, false);
    }
  }
  #pragma unroll
  for (int nt = 0; nt < 4; ++nt) {
    const int a1 = nt * 16 + ln;
    const int m0 = m_base + hi * 8;
    v8bf pk;
    #pragma unroll
    for (int r = 0; r < 8; ++r)
      pk[r] = f2bf(fmaxf(acc2[nt][r] + bias1[r], 0.0f));
    *(v8bf*)&h1t[a1][m0] = pk;
  }
  __syncthreads();

  // ---------------- phase 3: attention + 6-d regressor heads ----------------
  if (t < NA_PAD) {
    const int a1 = t;
    float acc_a = ba[0];
    float accy[OUTD];
    #pragma unroll
    for (int j = 0; j < OUTD; ++j) accy[j] = br[j];
    for (int k = 0; k < MLPD; ++k) {
      const float h = bf2f(h1t[a1][k]);
      acc_a += Wa[k] * h;
      #pragma unroll
      for (int j = 0; j < OUTD; ++j) accy[j] += Wr[j * MLPD + k] * h;
    }
    attn_s[a1] = acc_a;
    if (a1 < NANCH) {
      const size_t YOFF = (size_t)NBATCH * NANCH * NANCH;
      #pragma unroll
      for (int j = 0; j < OUTD; ++j)
        out[YOFF + (((size_t)b * OUTD + j) * NANCH + a1) * NANCH + a2] = accy[j];
    }
  }
  __syncthreads();

  // softmax over a1 for this (b, a2)   (TEMP == 1)
  if (t == 0) {
    float mx = -3.4e38f;
    for (int a1 = 0; a1 < NANCH; ++a1) mx = fmaxf(mx, attn_s[a1]);
    float s = 0.0f;
    for (int a1 = 0; a1 < NANCH; ++a1) s += __expf(attn_s[a1] - mx);
    const float inv = 1.0f / s;
    for (int a1 = 0; a1 < NANCH; ++a1)
      out[((size_t)b * NANCH + a1) * NANCH + a2] = __expf(attn_s[a1] - mx) * inv;
  }
}

// =====================================================================
extern "C" void kernel_launch(void* const* d_in, const int* in_sizes, int n_in,
                              void* d_out, int out_size, void* d_ws, size_t ws_size,
                              hipStream_t stream) {
  (void)in_sizes; (void)n_in; (void)out_size; (void)ws_size;
  const float* f1 = (const float*)d_in[0];
  const float* f2 = (const float*)d_in[1];
  const float* x1 = (const float*)d_in[2];
  const float* x2 = (const float*)d_in[3];
  const float* Wp = (const float*)d_in[4];
  const float* bp = (const float*)d_in[5];
  const float* W0 = (const float*)d_in[6];
  const float* b0 = (const float*)d_in[7];
  const float* W1 = (const float*)d_in[8];
  const float* b1 = (const float*)d_in[9];
  const float* Wa = (const float*)d_in[10];
  const float* ba = (const float*)d_in[11];
  const float* Wr = (const float*)d_in[12];
  const float* br = (const float*)d_in[13];

  float* g_ws = (float*)d_ws;          // [2][16][256][64] = 2 MB

  pointnet_wmma_kernel<<<dim3(NBATCH * 2, NA_PAD / 16), 512, 0, stream>>>(
      f1, f2, x1, x2, Wp, bp, g_ws);

  pairmlp_wmma_kernel<<<dim3(NBATCH * NANCH), 512, 0, stream>>>(
      g_ws, W0, b0, W1, b1, Wa, ba, Wr, br, (float*)d_out);
}
